// SchEmbedding_47614007443633
// MI455X (gfx1250) — matrix operand order repaired and checked by reading, not compiled, not run on hardware
//
#include <hip/hip_runtime.h>
#include <hip/hip_bf16.h>
#include <math.h>

// ---- static module config ----
#define NNODES   100000
#define NEDGES   1600000
#define NGRAPH   128
#define FDIM     64
#define NCENT    50
#define NCONV    3
#define RBF_GAP  (5.0f / 49.0f)
#define RBF_COEF (-49.0f / 5.0f)

#define TILES_PER_BLOCK 8                               // 16-row tiles per wave
#define EDGE_BLOCKS (NEDGES / (16 * TILES_PER_BLOCK))   // 12500 exactly
#define NODE_BLOCKS ((NNODES + 16 * TILES_PER_BLOCK - 1) / (16 * TILES_PER_BLOCK))  // 782
#define NPAD (NODE_BLOCKS * 16 * TILES_PER_BLOCK)       // 100096 padded rows

typedef __attribute__((ext_vector_type(16))) _Float16 v16h;
typedef __attribute__((ext_vector_type(8)))  float    v8f;
typedef __attribute__((ext_vector_type(4)))  float    v4f;

union F16x16 { v16h v; _Float16 h[16]; };
union F32x8  { v8f  v; float    f[8]; };
union F32x4  { v4f  v; float    f[4]; };

static __device__ __forceinline__ v8f wmma_f16(v16h a, v16h b, v8f c) {
  // D = A(16x32 f16) x B(32x16 f16) + C(16x16 f32)
  return __builtin_amdgcn_wmma_f32_16x16x32_f16(
      /*neg_a=*/false, a, /*neg_b=*/false, b,
      /*c_mod=*/(short)0, c, /*reuse_a=*/false, /*reuse_b=*/false);
}

static __device__ __forceinline__ float sp05f(float x) {
  float hx = 0.5f * x;
  if (hx > 14.0f) return x;
  return 2.0f * __logf(1.0f + __expf(hx));
}

static __device__ __forceinline__ float shiftspf(float x) {
  float sp = (x > 20.0f) ? x : __logf(1.0f + __expf(x));
  return sp - 0.6931471805599453f;
}

static __device__ __forceinline__ float rbff(float d, int k) {
  if (k >= NCENT) return 0.0f;
  float diff = d - (float)k * RBF_GAP;
  return __expf(RBF_COEF * diff * diff);
}

// gfx1250 async global->LDS copy: one wave op moves 16B/lane into LDS,
// tracked with ASYNCcnt (CDNA5 ISA).
static __device__ __forceinline__ void async_copy_b128(unsigned lds_addr,
                                                       const void* gaddr) {
  asm volatile("global_load_async_to_lds_b128 %0, %1, off"
               :: "v"(lds_addr), "v"((unsigned long long)gaddr)
               : "memory");
}
static __device__ __forceinline__ void wait_asynccnt0() {
  asm volatile("s_wait_asynccnt 0x0" ::: "memory");
}

// Build an A fragment pair (16x64 rows of f32 -> two 16x32 f16 fragments)
// from a contiguous 64-float row. lane: grp = lane>>4, row = lane&15.
static __device__ __forceinline__ void load_a_frags(const float* __restrict__ row,
                                                    int grp, F16x16& a0, F16x16& a1) {
  const v4f* r4 = (const v4f*)row;
  F32x4 c[8];
#pragma unroll
  for (int q = 0; q < 4; ++q) {
    c[2 * q].v     = r4[q * 4 + grp * 2];
    c[2 * q + 1].v = r4[q * 4 + grp * 2 + 1];
  }
#pragma unroll
  for (int j = 0; j < 8; ++j) {
    a0.h[j]     = (_Float16)c[0 + j / 4].f[j & 3];
    a0.h[j + 8] = (_Float16)c[2 + j / 4].f[j & 3];
    a1.h[j]     = (_Float16)c[4 + j / 4].f[j & 3];
    a1.h[j + 8] = (_Float16)c[6 + j / 4].f[j & 3];
  }
}

// ---------------------------------------------------------------------------
// Weight prep -> f16 B-fragment layout: [n_tile(4)][k_chunk(2)][lane(32)][j(16)]
//   elem j of lane L (g=L>>4, n=L&15): B[k_chunk*32 + (j<8 ? g*8+j : 16+g*8+j-8)][n_tile*16 + n]
// ---------------------------------------------------------------------------
__global__ __launch_bounds__(32) void prep_weights_kernel(
    const float* __restrict__ W1, const float* __restrict__ cfW1,
    const float* __restrict__ cfW2, const float* __restrict__ W2,
    const float* __restrict__ W3, const float* __restrict__ Wd1,
    _Float16* __restrict__ frag) {
  const int mat  = blockIdx.x;
  const int lane = threadIdx.x;
  const int g = lane >> 4, n0 = lane & 15;
  const float* src;
  int kmax = 64;
  if      (mat <  3) { src = W1   + mat * 4096; }
  else if (mat <  6) { src = cfW1 + (mat - 3) * (NCENT * 64); kmax = NCENT; }
  else if (mat <  9) { src = cfW2 + (mat - 6) * 4096; }
  else if (mat < 12) { src = W2   + (mat - 9) * 4096; }
  else if (mat < 15) { src = W3   + (mat - 12) * 4096; }
  else               { src = Wd1; }
  _Float16* out = frag + mat * 4096;
#pragma unroll
  for (int nt = 0; nt < 4; ++nt)
#pragma unroll
    for (int kc = 0; kc < 2; ++kc)
#pragma unroll
      for (int j = 0; j < 16; ++j) {
        int k = kc * 32 + (j < 8 ? g * 8 + j : 16 + g * 8 + (j - 8));
        int n = nt * 16 + n0;
        float v = (k < kmax) ? src[k * 64 + n] : 0.0f;
        out[((nt * 2 + kc) * 32 + lane) * 16 + j] = (_Float16)v;
      }
}

// node = emb[nodes]; pad rows -> 0
__global__ void embed_kernel(const int* __restrict__ nodes,
                             const float* __restrict__ emb,
                             float* __restrict__ node) {
  int idx = blockIdx.x * blockDim.x + threadIdx.x;
  if (idx < NPAD * FDIM) {
    int row = idx >> 6, col = idx & 63;
    node[idx] = (row < NNODES) ? emb[nodes[row] * FDIM + col] : 0.0f;
  }
}

__global__ void zero_graph_kernel(float* __restrict__ gacc) {
  int i = blockIdx.x * blockDim.x + threadIdx.x;
  if (i < 2 * NGRAPH) gacc[i] = 0.0f;
}

__global__ void count_kernel(const int* __restrict__ gids,
                             float* __restrict__ gcnt) {
  int i = blockIdx.x * blockDim.x + threadIdx.x;
  if (i < NNODES) atomicAdd(&gcnt[gids[i]], 1.0f);
}

__global__ void finalize_kernel(const float* __restrict__ gsum,
                                const float* __restrict__ gcnt,
                                const float* __restrict__ bd2,
                                float* __restrict__ out) {
  int g = blockIdx.x * blockDim.x + threadIdx.x;
  if (g < NGRAPH) out[g] = gsum[g] / gcnt[g] + bd2[0];
}

// ---------------------------------------------------------------------------
// new_node = node @ W1[l]  (no bias), also zero cf. Rows padded -> no guards.
// ---------------------------------------------------------------------------
__global__ __launch_bounds__(32) void node_gemm_kernel(
    const float* __restrict__ nodeIn, const _Float16* __restrict__ fragW1,
    float* __restrict__ new_node, float* __restrict__ cfZ) {
  __shared__ v16h sB[8][32];
  const int lane = threadIdx.x;
  const int grp = lane >> 4, ln = lane & 15;
  {
    const v16h* gW = (const v16h*)fragW1;
    v16h* p = &sB[0][0];
#pragma unroll
    for (int i = 0; i < 8; ++i) p[i * 32 + lane] = gW[i * 32 + lane];
  }
  __syncthreads();
  for (int t = 0; t < TILES_PER_BLOCK; ++t) {
    const int rbase = (blockIdx.x * TILES_PER_BLOCK + t) * 16;
    F16x16 a0, a1;
    load_a_frags(nodeIn + (size_t)(rbase + ln) * FDIM, grp, a0, a1);
#pragma unroll
    for (int nt = 0; nt < 4; ++nt) {
      F32x8 acc;
#pragma unroll
      for (int i = 0; i < 8; ++i) acc.f[i] = 0.0f;
      acc.v = wmma_f16(a0.v, sB[nt * 2 + 0][lane], acc.v);
      acc.v = wmma_f16(a1.v, sB[nt * 2 + 1][lane], acc.v);
      const int col = nt * 16 + ln;
#pragma unroll
      for (int v = 0; v < 8; ++v) {
        int row = rbase + v + grp * 8;
        new_node[row * FDIM + col] = acc.f[v];
        cfZ[row * FDIM + col] = 0.0f;
      }
    }
  }
}

// ---------------------------------------------------------------------------
// Edge filter + message + scatter:
//   h = sp05(rbf @ cfW1 + cfb1) @ cfW2 + cfb2 ; cf[dst] += new_node[src] * h
// ---------------------------------------------------------------------------
__global__ __launch_bounds__(32) void edge_kernel(
    const float* __restrict__ distance, const int* __restrict__ src,
    const int* __restrict__ dst, const float* __restrict__ new_node,
    const float* __restrict__ cfb1, const float* __restrict__ cfb2,
    const _Float16* __restrict__ fragCf1, const _Float16* __restrict__ fragCf2,
    float* __restrict__ cf) {
  __shared__ v16h sB1[8][32];
  __shared__ v16h sB2[8][32];
  __shared__ _Float16 stage[16 * FDIM];
  const int lane = threadIdx.x;
  const int grp = lane >> 4, ln = lane & 15;
  {
    // async global->LDS DMA of both weight matrices (16KB total),
    // 16B/lane per instruction, per-lane LDS byte addresses.
    unsigned l1 = (unsigned)(size_t)(&sB1[0][0]) + lane * 16u;
    unsigned l2 = (unsigned)(size_t)(&sB2[0][0]) + lane * 16u;
    const char* g1 = (const char*)fragCf1 + lane * 16;
    const char* g2 = (const char*)fragCf2 + lane * 16;
#pragma unroll
    for (int i = 0; i < 16; ++i) {
      async_copy_b128(l1 + i * 512u, g1 + i * 512);
      async_copy_b128(l2 + i * 512u, g2 + i * 512);
    }
    wait_asynccnt0();
  }
  __syncthreads();
  // hoisted per-lane bias values
  float cb1[4], cb2[4];
#pragma unroll
  for (int nt = 0; nt < 4; ++nt) {
    cb1[nt] = cfb1[nt * 16 + ln];
    cb2[nt] = cfb2[nt * 16 + ln];
  }
  for (int t = 0; t < TILES_PER_BLOCK; ++t) {
    const int ebase = (blockIdx.x * TILES_PER_BLOCK + t) * 16;
    if (t + 1 < TILES_PER_BLOCK)
      __builtin_prefetch(&distance[ebase + 16], 0, 1);
    const float d = distance[ebase + ln];
    // RBF straight into A-fragment registers (16x64, K zero-padded)
    F16x16 a0, a1;
#pragma unroll
    for (int j = 0; j < 8; ++j) {
      a0.h[j]     = (_Float16)rbff(d, grp * 8 + j);
      a0.h[j + 8] = (_Float16)rbff(d, 16 + grp * 8 + j);
      a1.h[j]     = (_Float16)rbff(d, 32 + grp * 8 + j);
      a1.h[j + 8] = (_Float16)rbff(d, 48 + grp * 8 + j);
    }
    int sE[8], dE[8];
#pragma unroll
    for (int v = 0; v < 8; ++v) {
      int e = ebase + v + grp * 8;
      sE[v] = src[e];
      dE[v] = dst[e];
    }
    // GEMM1: rbf @ cfW1 ; softplus ; stage for relayout
#pragma unroll
    for (int nt = 0; nt < 4; ++nt) {
      F32x8 acc;
#pragma unroll
      for (int i = 0; i < 8; ++i) acc.f[i] = 0.0f;
      acc.v = wmma_f16(a0.v, sB1[nt * 2 + 0][lane], acc.v);
      acc.v = wmma_f16(a1.v, sB1[nt * 2 + 1][lane], acc.v);
      const int col = nt * 16 + ln;
#pragma unroll
      for (int v = 0; v < 8; ++v)
        stage[(v + grp * 8) * FDIM + col] = (_Float16)sp05f(acc.f[v] + cb1[nt]);
    }
    __syncthreads();
    F16x16 t0, t1;
    {
      const _Float16* srow = &stage[ln * FDIM];
#pragma unroll
      for (int j = 0; j < 8; ++j) {
        t0.h[j]     = srow[grp * 8 + j];
        t0.h[j + 8] = srow[16 + grp * 8 + j];
        t1.h[j]     = srow[32 + grp * 8 + j];
        t1.h[j + 8] = srow[48 + grp * 8 + j];
      }
    }
    __syncthreads();
    // GEMM2: @ cfW2 ; gather-multiply-scatter from C layout
#pragma unroll
    for (int nt = 0; nt < 4; ++nt) {
      F32x8 acc;
#pragma unroll
      for (int i = 0; i < 8; ++i) acc.f[i] = 0.0f;
      acc.v = wmma_f16(t0.v, sB2[nt * 2 + 0][lane], acc.v);
      acc.v = wmma_f16(t1.v, sB2[nt * 2 + 1][lane], acc.v);
      const int col = nt * 16 + ln;
#pragma unroll
      for (int v = 0; v < 8; ++v) {
        float h = acc.f[v] + cb2[nt];
        float msg = new_node[sE[v] * FDIM + col] * h;
        atomicAdd(&cf[dE[v] * FDIM + col], msg);
      }
    }
  }
}

// ---------------------------------------------------------------------------
// node += sp05(cf @ W2 + b2) @ W3 + b3   (rows padded -> no guards)
// ---------------------------------------------------------------------------
__global__ __launch_bounds__(32) void update_kernel(
    const float* __restrict__ cf, const float* __restrict__ b2,
    const float* __restrict__ b3, const _Float16* __restrict__ fragW2,
    const _Float16* __restrict__ fragW3, float* __restrict__ node) {
  __shared__ v16h sB2[8][32];
  __shared__ v16h sB3[8][32];
  __shared__ _Float16 stage[16 * FDIM];
  const int lane = threadIdx.x;
  const int grp = lane >> 4, ln = lane & 15;
  {
    const v16h* g2 = (const v16h*)fragW2;
    const v16h* g3 = (const v16h*)fragW3;
    v16h* p2 = &sB2[0][0];
    v16h* p3 = &sB3[0][0];
#pragma unroll
    for (int i = 0; i < 8; ++i) {
      p2[i * 32 + lane] = g2[i * 32 + lane];
      p3[i * 32 + lane] = g3[i * 32 + lane];
    }
  }
  __syncthreads();
  float pb2[4], pb3[4];
#pragma unroll
  for (int nt = 0; nt < 4; ++nt) {
    pb2[nt] = b2[nt * 16 + ln];
    pb3[nt] = b3[nt * 16 + ln];
  }
  for (int t = 0; t < TILES_PER_BLOCK; ++t) {
    const int rbase = (blockIdx.x * TILES_PER_BLOCK + t) * 16;
    F16x16 a0, a1;
    load_a_frags(cf + (size_t)(rbase + ln) * FDIM, grp, a0, a1);
#pragma unroll
    for (int nt = 0; nt < 4; ++nt) {
      F32x8 acc;
#pragma unroll
      for (int i = 0; i < 8; ++i) acc.f[i] = 0.0f;
      acc.v = wmma_f16(a0.v, sB2[nt * 2 + 0][lane], acc.v);
      acc.v = wmma_f16(a1.v, sB2[nt * 2 + 1][lane], acc.v);
      const int col = nt * 16 + ln;
#pragma unroll
      for (int v = 0; v < 8; ++v)
        stage[(v + grp * 8) * FDIM + col] = (_Float16)sp05f(acc.f[v] + pb2[nt]);
    }
    __syncthreads();
    F16x16 t0, t1;
    {
      const _Float16* srow = &stage[ln * FDIM];
#pragma unroll
      for (int j = 0; j < 8; ++j) {
        t0.h[j]     = srow[grp * 8 + j];
        t0.h[j + 8] = srow[16 + grp * 8 + j];
        t1.h[j]     = srow[32 + grp * 8 + j];
        t1.h[j + 8] = srow[48 + grp * 8 + j];
      }
    }
    __syncthreads();
#pragma unroll
    for (int nt = 0; nt < 4; ++nt) {
      F32x8 acc;
#pragma unroll
      for (int i = 0; i < 8; ++i) acc.f[i] = 0.0f;
      acc.v = wmma_f16(t0.v, sB3[nt * 2 + 0][lane], acc.v);
      acc.v = wmma_f16(t1.v, sB3[nt * 2 + 1][lane], acc.v);
      const int col = nt * 16 + ln;
#pragma unroll
      for (int v = 0; v < 8; ++v) {
        int row = rbase + v + grp * 8;
        node[row * FDIM + col] += acc.f[v] + pb3[nt];
      }
    }
  }
}

// ---------------------------------------------------------------------------
// Readout: res = shiftsp(node @ Wd1 + bd1); per-node scalar = res @ Wd2;
// atomic per-graph accumulate (mean@Wd2 == mean(res@Wd2) by linearity)
// ---------------------------------------------------------------------------
__global__ __launch_bounds__(32) void final_kernel(
    const float* __restrict__ node, const float* __restrict__ bd1,
    const float* __restrict__ Wd2, const int* __restrict__ gids,
    const _Float16* __restrict__ fragWd1, float* __restrict__ gsum) {
  __shared__ v16h sB[8][32];
  const int lane = threadIdx.x;
  const int grp = lane >> 4, ln = lane & 15;
  {
    const v16h* gW = (const v16h*)fragWd1;
    v16h* p = &sB[0][0];
#pragma unroll
    for (int i = 0; i < 8; ++i) p[i * 32 + lane] = gW[i * 32 + lane];
  }
  __syncthreads();
  float pb[4], pw[4];
#pragma unroll
  for (int nt = 0; nt < 4; ++nt) {
    pb[nt] = bd1[nt * 16 + ln];
    pw[nt] = Wd2[nt * 16 + ln];
  }
  for (int t = 0; t < TILES_PER_BLOCK; ++t) {
    const int rbase = (blockIdx.x * TILES_PER_BLOCK + t) * 16;
    F16x16 a0, a1;
    load_a_frags(node + (size_t)(rbase + ln) * FDIM, grp, a0, a1);
#pragma unroll
    for (int nt = 0; nt < 4; ++nt) {
      F32x8 acc;
#pragma unroll
      for (int i = 0; i < 8; ++i) acc.f[i] = 0.0f;
      acc.v = wmma_f16(a0.v, sB[nt * 2 + 0][lane], acc.v);
      acc.v = wmma_f16(a1.v, sB[nt * 2 + 1][lane], acc.v);
#pragma unroll
      for (int v = 0; v < 8; ++v) {
        int row = rbase + v + grp * 8;
        if (row < NNODES) {
          float r = shiftspf(acc.f[v] + pb[nt]);
          atomicAdd(&gsum[gids[row]], r * pw[nt]);
        }
      }
    }
  }
}

// ---------------------------------------------------------------------------
extern "C" void kernel_launch(void* const* d_in, const int* in_sizes, int n_in,
                              void* d_out, int out_size, void* d_ws, size_t ws_size,
                              hipStream_t stream) {
  (void)in_sizes; (void)n_in; (void)out_size; (void)ws_size;
  const int*   nodes    = (const int*)d_in[0];
  const float* distance = (const float*)d_in[1];
  const int*   src      = (const int*)d_in[2];
  const int*   dst      = (const int*)d_in[3];
  const int*   gids     = (const int*)d_in[4];
  // d_in[5] = n_graphs (statically NGRAPH)
  const float* emb  = (const float*)d_in[6];
  const float* W1   = (const float*)d_in[7];
  const float* cfW1 = (const float*)d_in[8];
  const float* cfb1 = (const float*)d_in[9];
  const float* cfW2 = (const float*)d_in[10];
  const float* cfb2 = (const float*)d_in[11];
  const float* W2   = (const float*)d_in[12];
  const float* b2   = (const float*)d_in[13];
  const float* W3   = (const float*)d_in[14];
  const float* b3   = (const float*)d_in[15];
  const float* Wd1  = (const float*)d_in[16];
  const float* bd1  = (const float*)d_in[17];
  const float* Wd2  = (const float*)d_in[18];
  const float* bd2  = (const float*)d_in[19];
  float* out = (float*)d_out;

  // workspace layout (node arrays padded to NPAD rows)
  float* node     = (float*)d_ws;
  float* new_node = node + (size_t)NPAD * FDIM;
  float* cf       = new_node + (size_t)NPAD * FDIM;
  _Float16* frag  = (_Float16*)(cf + (size_t)NPAD * FDIM);  // 16 * 4096 halves
  float* gsum     = (float*)(frag + 16 * 4096);
  float* gcnt     = gsum + NGRAPH;

  prep_weights_kernel<<<16, 32, 0, stream>>>(W1, cfW1, cfW2, W2, W3, Wd1, frag);
  embed_kernel<<<(NPAD * FDIM + 255) / 256, 256, 0, stream>>>(nodes, emb, node);
  zero_graph_kernel<<<1, 2 * NGRAPH, 0, stream>>>(gsum);

  for (int l = 0; l < NCONV; ++l) {
    node_gemm_kernel<<<NODE_BLOCKS, 32, 0, stream>>>(
        node, frag + (0 + l) * 4096, new_node, cf);
    edge_kernel<<<EDGE_BLOCKS, 32, 0, stream>>>(
        distance, src, dst, new_node, cfb1 + l * FDIM, cfb2 + l * FDIM,
        frag + (3 + l) * 4096, frag + (6 + l) * 4096, cf);
    update_kernel<<<NODE_BLOCKS, 32, 0, stream>>>(
        cf, b2 + l * FDIM, b3 + l * FDIM,
        frag + (9 + l) * 4096, frag + (12 + l) * 4096, node);
  }

  count_kernel<<<(NNODES + 255) / 256, 256, 0, stream>>>(gids, gcnt);
  final_kernel<<<NODE_BLOCKS, 32, 0, stream>>>(node, bd1, Wd2, gids,
                                               frag + 15 * 4096, gsum);
  finalize_kernel<<<1, NGRAPH, 0, stream>>>(gsum, gcnt, bd2, out);
}
